// HMULayer_omega_49520972923447
// MI455X (gfx1250) — matrix operand order repaired and checked by reading, not compiled, not run on hardware
//
#include <hip/hip_runtime.h>
#include <hip/hip_bf16.h>

// ---------------- problem constants ----------------
#define B_SZ 1024
#define N_SZ 8192
#define D_SZ 256
#define K_SZ 8

#define BT 128   // b rows per workgroup
#define NT 16    // n cols per workgroup
#define DK 32    // depth per WMMA step (bf16 16x16x32)
#define ROWP 40  // LDS row pitch in bf16 elements (80B, 16B aligned, bank-friendly)

typedef __attribute__((ext_vector_type(16))) __bf16 v16bf;
typedef __attribute__((ext_vector_type(8)))  float  v8f;
typedef __attribute__((ext_vector_type(4)))  unsigned int u32x4;
typedef __attribute__((ext_vector_type(2)))  unsigned int u32x2;

union V16 { u32x4 q[2]; v16bf v; };

// ---------------- helpers ----------------
__device__ __forceinline__ unsigned short bf_rne(float f) {
  unsigned int u = __float_as_uint(f);
  unsigned int r = u + 0x7FFFu + ((u >> 16) & 1u);
  return (unsigned short)(r >> 16);
}
__device__ __forceinline__ float bf_f(unsigned short h) {
  return __uint_as_float(((unsigned int)h) << 16);
}

// split 4 consecutive f32 into bf16 hi/lo and store to LDS (8B stores)
__device__ __forceinline__ void stage4(unsigned short* hiA, unsigned short* loA,
                                       int idx, float4 f) {
  float a0 = f.x, a1 = f.y, a2 = f.z, a3 = f.w;
  unsigned short h0 = bf_rne(a0), h1 = bf_rne(a1), h2 = bf_rne(a2), h3 = bf_rne(a3);
  unsigned short l0 = bf_rne(a0 - bf_f(h0));
  unsigned short l1 = bf_rne(a1 - bf_f(h1));
  unsigned short l2 = bf_rne(a2 - bf_f(h2));
  unsigned short l3 = bf_rne(a3 - bf_f(h3));
  u32x2 hp, lp;
  hp.x = (unsigned int)h0 | ((unsigned int)h1 << 16);
  hp.y = (unsigned int)h2 | ((unsigned int)h3 << 16);
  lp.x = (unsigned int)l0 | ((unsigned int)l1 << 16);
  lp.y = (unsigned int)l2 | ((unsigned int)l3 << 16);
  *(u32x2*)(hiA + idx) = hp;
  *(u32x2*)(loA + idx) = lp;
}

__device__ __forceinline__ float wave_reduce_add(float s) {
  s += __shfl_xor(s, 16, 32);
  s += __shfl_xor(s, 8, 32);
  s += __shfl_xor(s, 4, 32);
  s += __shfl_xor(s, 2, 32);
  s += __shfl_xor(s, 1, 32);
  return s;
}

// ---------------- precompute: x2[b] = ||x_b||^2 (one wave per b) ----------------
__global__ __launch_bounds__(256) void hmu_x2_kernel(const float* __restrict__ x,
                                                     float* __restrict__ x2) {
  int gw = (blockIdx.x * blockDim.x + threadIdx.x) >> 5;  // wave id == b
  int lane = threadIdx.x & 31;
  if (gw >= B_SZ) return;
  const float* row = x + (size_t)gw * D_SZ;
  float s = 0.f;
#pragma unroll
  for (int i = 0; i < D_SZ / 32; ++i) {
    float t = row[lane + 32 * i];
    s += t * t;
  }
  s = wave_reduce_add(s);
  if (lane == 0) x2[gw] = s;
}

// ------------- precompute: mu2[n], muv[n][k] = mu_n . v_nk (one wave per n) -------------
__global__ __launch_bounds__(256) void hmu_mustats_kernel(const float* __restrict__ mu,
                                                          const float* __restrict__ v,
                                                          float* __restrict__ mu2,
                                                          float* __restrict__ muv) {
  int n = (blockIdx.x * blockDim.x + threadIdx.x) >> 5;
  int lane = threadIdx.x & 31;
  if (n >= N_SZ) return;
  const float* mrow = mu + (size_t)n * D_SZ;
  float m[D_SZ / 32];
#pragma unroll
  for (int i = 0; i < D_SZ / 32; ++i) m[i] = mrow[lane + 32 * i];
  float s = 0.f;
#pragma unroll
  for (int i = 0; i < D_SZ / 32; ++i) s += m[i] * m[i];
  s = wave_reduce_add(s);
  if (lane == 0) mu2[n] = s;
#pragma unroll
  for (int k = 0; k < K_SZ; ++k) {
    const float* vr = v + ((size_t)n * K_SZ + k) * D_SZ;
    float t = 0.f;
#pragma unroll
    for (int i = 0; i < D_SZ / 32; ++i) t += m[i] * vr[lane + 32 * i];
    t = wave_reduce_add(t);
    if (lane == 0) muv[n * K_SZ + k] = t;
  }
}

// ---------------- main fused kernel ----------------
// grid = (N/NT, B/BT), block = 256 (8 waves). Wave w owns b rows [w*16, w*16+16)
// of the workgroup's 128-row slab, and all 9 column tiles (tile 0 = mu, 1..8 = v).
__global__ __launch_bounds__(256) void hmu_main_kernel(
    const float* __restrict__ x, const float* __restrict__ mu,
    const float* __restrict__ lambda_base, const float* __restrict__ v,
    const float* __restrict__ omega, const float* __restrict__ x2W,
    const float* __restrict__ mu2W, const float* __restrict__ muvW,
    float* __restrict__ out) {
  __shared__ __align__(16) unsigned short s_x_hi[BT * ROWP];
  __shared__ __align__(16) unsigned short s_x_lo[BT * ROWP];
  __shared__ __align__(16) unsigned short s_w_hi[(NT + NT * K_SZ) * ROWP];  // 144 rows
  __shared__ __align__(16) unsigned short s_w_lo[(NT + NT * K_SZ) * ROWP];

  const int tid = threadIdx.x;
  const int n0 = blockIdx.x * NT;
  const int b0 = blockIdx.y * BT;
  const int lane = tid & 31;
  const int w = tid >> 5;
  const int half = lane >> 4;   // which 16-lane half
  const int coln = lane & 15;   // WMMA column / row-within-tile

  v8f acc[9];
#pragma unroll
  for (int t = 0; t < 9; ++t)
#pragma unroll
    for (int r = 0; r < 8; ++r) acc[t][r] = 0.f;

  for (int dstep = 0; dstep < D_SZ / DK; ++dstep) {
    const int d0 = dstep * DK;

    // ---- stage x tile: 128 rows x 32 d, f32 -> bf16 hi/lo ----
#pragma unroll
    for (int it = 0; it < 4; ++it) {
      int q = tid + it * 256;             // 1024 float4s total
      int row = q >> 3;
      int c4 = (q & 7) << 2;
      const float* src = x + (size_t)(b0 + row) * D_SZ + d0 + c4;
      float4 f = *(const float4*)src;
      stage4(s_x_hi, s_x_lo, row * ROWP + c4, f);
      if (dstep < 7) __builtin_prefetch(src + DK, 0, 1);
    }
    // ---- stage W tile: rows 0..15 = mu[n0+..], rows 16..143 = v[n0*8 + ..] ----
    for (int q = tid; q < 144 * 8; q += 256) {
      int row = q >> 3;
      int c4 = (q & 7) << 2;
      const float* src = (row < NT)
                             ? (mu + (size_t)(n0 + row) * D_SZ + d0 + c4)
                             : (v + (size_t)(n0 * K_SZ + (row - NT)) * D_SZ + d0 + c4);
      float4 f = *(const float4*)src;
      stage4(s_w_hi, s_w_lo, row * ROWP + c4, f);
      if (dstep < 7) __builtin_prefetch(src + DK, 0, 1);
    }
    __syncthreads();

    // ---- A operand (x tile): 16x32 bf16. Lane holds M=coln; K runs at
    //      8*half and 16+8*half (per ISA 16-bit A layout). Two b128 loads each. ----
    V16 a_hi, a_lo;
    {
      int abase = (w * 16 + coln) * ROWP + 8 * half;
      a_hi.q[0] = *(const u32x4*)&s_x_hi[abase];
      a_hi.q[1] = *(const u32x4*)&s_x_hi[abase + 16];
      a_lo.q[0] = *(const u32x4*)&s_x_lo[abase];
      a_lo.q[1] = *(const u32x4*)&s_x_lo[abase + 16];
    }
#pragma unroll
    for (int t = 0; t < 9; ++t) {
      // B operand (W rows as columns): lane holds N=coln -> W row t*16+coln,
      // K runs at 8*half and 16+8*half (matching A layout).
      V16 b_hi, b_lo;
      int bbase = (t * 16 + coln) * ROWP + 8 * half;
      b_hi.q[0] = *(const u32x4*)&s_w_hi[bbase];
      b_hi.q[1] = *(const u32x4*)&s_w_hi[bbase + 16];
      b_lo.q[0] = *(const u32x4*)&s_w_lo[bbase];
      b_lo.q[1] = *(const u32x4*)&s_w_lo[bbase + 16];
      // bf16x3 split: hi*lo + lo*hi + hi*hi  (lo*lo dropped, ~2^-18 rel)
      acc[t] = __builtin_amdgcn_wmma_f32_16x16x32_bf16(false, a_lo.v, false, b_hi.v,
                                                       (short)0, acc[t], false, false);
      acc[t] = __builtin_amdgcn_wmma_f32_16x16x32_bf16(false, a_hi.v, false, b_lo.v,
                                                       (short)0, acc[t], false, false);
      acc[t] = __builtin_amdgcn_wmma_f32_16x16x32_bf16(false, a_hi.v, false, b_hi.v,
                                                       (short)0, acc[t], false, false);
    }
    __syncthreads();
  }

  // ---------------- epilogue ----------------
  // C/D layout: VGPR r holds M = r + 8*half, N = coln.
  // acc[0] = xm tile (cols = n0+coln). acc[t],t>=1: col -> v row (t-1)*16+coln,
  // i.e. n = n0 + 2(t-1) + (coln>>3), k = coln&7  (k lives in lane bits 0..2).
  float x2v[8];
#pragma unroll
  for (int r = 0; r < 8; ++r) x2v[r] = x2W[b0 + w * 16 + r + 8 * half];

#pragma unroll
  for (int t = 1; t <= 8; ++t) {
    const int nt = n0 + 2 * (t - 1) + (coln >> 3);
    const int kk = coln & 7;
    const float om = omega[nt * K_SZ + kk];
    const float mv = muvW[nt * K_SZ + kk];
    const float lam = lambda_base[nt];
    const float m2 = mu2W[nt];
    const int srcl = (half << 4) | (2 * (t - 1) + (coln >> 3));  // lane holding xm col nt
#pragma unroll
    for (int r = 0; r < 8; ++r) {
      float pk = acc[t][r] - mv;
      float s = om * pk * pk;
      s += __shfl_xor(s, 1, 32);   // reduce over k (lane bits 0..2)
      s += __shfl_xor(s, 2, 32);
      s += __shfl_xor(s, 4, 32);
      float xmv = __shfl(acc[0][r], srcl, 32);
      float quad = lam * (x2v[r] - 2.f * xmv + m2) + s;
      float outv = __expf(-quad * (1.0f / 256.0f));
      if (kk == r) {  // one writer per 8-lane k-group per row slot
        int b = b0 + w * 16 + r + 8 * half;
        out[(size_t)b * N_SZ + nt] = outv;
      }
    }
  }
}

// ---------------- host launcher ----------------
extern "C" void kernel_launch(void* const* d_in, const int* in_sizes, int n_in,
                              void* d_out, int out_size, void* d_ws, size_t ws_size,
                              hipStream_t stream) {
  (void)in_sizes; (void)n_in; (void)out_size; (void)ws_size;
  const float* x   = (const float*)d_in[0];
  const float* mu  = (const float*)d_in[1];
  const float* lam = (const float*)d_in[2];
  const float* v   = (const float*)d_in[3];
  const float* om  = (const float*)d_in[4];
  float* out = (float*)d_out;

  float* ws   = (float*)d_ws;
  float* x2W  = ws;                        // 1024 f32
  float* mu2W = ws + B_SZ;                 // 8192 f32
  float* muvW = ws + B_SZ + N_SZ;          // 65536 f32  (total 292 KB)

  hmu_x2_kernel<<<dim3(B_SZ * 32 / 256), dim3(256), 0, stream>>>(x, x2W);
  hmu_mustats_kernel<<<dim3(N_SZ * 32 / 256), dim3(256), 0, stream>>>(mu, v, mu2W, muvW);
  hmu_main_kernel<<<dim3(N_SZ / NT, B_SZ / BT), dim3(256), 0, stream>>>(
      x, mu, lam, v, om, x2W, mu2W, muvW, out);
}
